// PointObSfcEncoder_65678639891298
// MI455X (gfx1250) — matrix operand order repaired and checked by reading, not compiled, not run on hardware
//
#include <hip/hip_runtime.h>
#include <hip/hip_bf16.h>

typedef __attribute__((ext_vector_type(16))) _Float16 v16h;
typedef __attribute__((ext_vector_type(8)))  _Float16 v8h;
typedef __attribute__((ext_vector_type(8)))  float    v8f;
typedef __attribute__((ext_vector_type(4)))  unsigned int v4u;
typedef __attribute__((ext_vector_type(4)))  int      v4i;
typedef __attribute__((ext_vector_type(8)))  int      v8i;

#define N_PTS   65534
#define N_VARS  24
#define GH      180
#define GW      360
#define CL      512
#define TR_DIM  128
#define TR_TOK  4
#define HEADS   8
#define DH      16
#define PE_NUM  8
#define EMB_IN  75
#define EMB_K   96
#define NTOK    (N_PTS*TR_TOK)
#define PI_F    3.14159265358979f

#if __has_builtin(__builtin_amdgcn_tensor_load_to_lds) && __has_builtin(__builtin_amdgcn_tensor_store_from_lds)
#define HAVE_TDM 1
#else
#define HAVE_TDM 0
#endif

// ================= TDM (Tensor Data Mover) helpers =================
#if HAVE_TDM
// Build D# groups 0/1 for a 2D f32 tile: tile0 x tile1 elements, tensor dim0 x dim1,
// row stride stride0 (elements). OOB rows: loads return 0, stores dropped.
__device__ __forceinline__ void tdm_desc_2d(v4u& g0, v8i& g1, unsigned lds_off,
                                            unsigned long long gaddr,
                                            unsigned dim0, unsigned dim1,
                                            unsigned tile0, unsigned tile1,
                                            unsigned long long stride0) {
  g0[0] = 1u;                                            // count=1, user desc
  g0[1] = lds_off;                                       // lds_addr
  g0[2] = (unsigned)(gaddr & 0xffffffffu);               // global_addr[31:0]
  g0[3] = (unsigned)((gaddr >> 32) & 0x01ffffffu) | (2u << 30);  // addr[56:32] | type=2
  g1[0] = (int)(2u << 16);                               // data_size=4B, mask=0
  g1[1] = (int)((dim0 & 0xffffu) << 16);                 // tensor_dim0[15:0]
  g1[2] = (int)(((dim0 >> 16) & 0xffffu) | ((dim1 & 0xffffu) << 16));
  g1[3] = (int)(((dim1 >> 16) & 0xffffu) | ((tile0 & 0xffffu) << 16));
  g1[4] = (int)(tile1 & 0xffffu);                        // tile_dim1, tile_dim2=0
  g1[5] = (int)(stride0 & 0xffffffffu);                  // dim0_stride[31:0]
  g1[6] = (int)((stride0 >> 32) & 0xffffu);              // dim0_stride[47:32]
  g1[7] = 0;
}

__device__ __forceinline__ void tdm_load_2d(void* lds_ptr, const void* gptr,
                                            unsigned dim0, unsigned dim1,
                                            unsigned tile0, unsigned tile1,
                                            unsigned long long stride0) {
  v4u g0; v8i g1;
  tdm_desc_2d(g0, g1, (unsigned)(uintptr_t)lds_ptr, (unsigned long long)(uintptr_t)gptr,
              dim0, dim1, tile0, tile1, stride0);
  v4i z4 = {0, 0, 0, 0};
#if __clang_major__ >= 23
  v8i z8 = {0, 0, 0, 0, 0, 0, 0, 0};
  __builtin_amdgcn_tensor_load_to_lds(g0, g1, z4, z4, z8, 0);
#else
  __builtin_amdgcn_tensor_load_to_lds(g0, g1, z4, z4, 0);
#endif
}

__device__ __forceinline__ void tdm_store_2d(void* lds_ptr, void* gptr,
                                             unsigned dim0, unsigned dim1,
                                             unsigned tile0, unsigned tile1,
                                             unsigned long long stride0) {
  v4u g0; v8i g1;
  tdm_desc_2d(g0, g1, (unsigned)(uintptr_t)lds_ptr, (unsigned long long)(uintptr_t)gptr,
              dim0, dim1, tile0, tile1, stride0);
  v4i z4 = {0, 0, 0, 0};
#if __clang_major__ >= 23
  v8i z8 = {0, 0, 0, 0, 0, 0, 0, 0};
  __builtin_amdgcn_tensor_store_from_lds(g0, g1, z4, z4, z8, 0);
#else
  __builtin_amdgcn_tensor_store_from_lds(g0, g1, z4, z4, 0);
#endif
}
#endif // HAVE_TDM

// ================= WMMA fragment helpers (wave32, 16x16x32 f16) =================
__device__ __forceinline__ v16h frag_a_lds(const _Float16* a, int lda, int k0) {
  int lane = threadIdx.x & 31;
  int m = lane & 15, hh = lane >> 4;
  const _Float16* row = a + m * lda + k0 + 8 * hh;
  v8h lo = *(const v8h*)row;
  v8h hi = *(const v8h*)(row + 16);
  return __builtin_shufflevector(lo, hi, 0,1,2,3,4,5,6,7,8,9,10,11,12,13,14,15);
}

// B packed layout: [K/32][N/16][32 lanes][16 f16]
__device__ __forceinline__ v16h frag_b_packed(const _Float16* bp, int NT, int kt, int nt) {
  int lane = threadIdx.x & 31;
  return *(const v16h*)(bp + ((((size_t)kt * NT + nt) * 32 + lane) << 4));
}

__device__ __forceinline__ void prefetch_b(const _Float16* bp, int NT, int kt, int nt) {
  int lane = threadIdx.x & 31;
  __builtin_prefetch(bp + ((((size_t)kt * NT + nt) * 32 + lane) << 4), 0, 3);
}

__device__ __forceinline__ v8f wmma32(v16h a, v16h b, v8f c) {
  return __builtin_amdgcn_wmma_f32_16x16x32_f16(false, a, false, b, (short)0, c, false, false);
}

__device__ __forceinline__ float gelu_tanh(float x) {
  return 0.5f * x * (1.0f + tanhf(0.7978845608f * (x + 0.044715f * x * x * x)));
}

// LN over 16 rows x 128 cols: 16 threads/row, shfl_xor reduction (wave32, width 16).
__device__ __forceinline__ void layernorm16(const float (*src)[TR_DIM], _Float16 (*dst)[TR_DIM],
                                            const float* __restrict__ s,
                                            const float* __restrict__ b, int tid) {
  int row = tid >> 4, sub = tid & 15;
  float vals[8];
  float sum = 0.f, sumsq = 0.f;
#pragma unroll
  for (int j = 0; j < 8; ++j) {
    float v = src[row][sub * 8 + j];
    vals[j] = v; sum += v; sumsq += v * v;
  }
#pragma unroll
  for (int m = 1; m < 16; m <<= 1) {
    sum   += __shfl_xor(sum, m, 16);
    sumsq += __shfl_xor(sumsq, m, 16);
  }
  float mean = sum * (1.0f / TR_DIM);
  float var  = sumsq * (1.0f / TR_DIM) - mean * mean;
  float rs = rsqrtf(var + 1e-5f);
#pragma unroll
  for (int j = 0; j < 8; ++j) {
    int c = sub * 8 + j;
    dst[row][c] = (_Float16)((vals[j] - mean) * rs * s[c] + b[c]);
  }
}

// ================= prep kernels =================
__global__ void k_pack_b(_Float16* __restrict__ dst, const float* __restrict__ src,
                         int K, int N, int Ksrc) {
  int i = blockIdx.x * blockDim.x + threadIdx.x;
  if (i >= K * N) return;
  int e    = i & 15;
  int lane = (i >> 4) & 31;
  int tile = i >> 9;
  int NT = N >> 4;
  int kt = tile / NT, nt = tile - kt * NT;
  int k = kt * 32 + ((lane >> 4) << 3) + (e & 7) + ((e >> 3) << 4);
  int n = nt * 16 + (lane & 15);
  dst[i] = (k < Ksrc) ? (_Float16)src[(size_t)k * N + n] : (_Float16)0.0f;
}

__global__ void k_zero(float* __restrict__ p, int n) {
  for (int i = blockIdx.x * blockDim.x + threadIdx.x; i < n; i += gridDim.x * blockDim.x)
    p[i] = 0.0f;
}

// ================= embed: feat -> h = feat @ W_emb + b =================
__global__ __launch_bounds__(256) void k_embed(const float* __restrict__ x,
                                               const _Float16* __restrict__ Wemb,
                                               const float* __restrict__ bemb,
                                               float* __restrict__ hbuf,
                                               int* __restrict__ flat) {
  __shared__ __align__(32) _Float16 sfeat[16][EMB_K];
  __shared__ __align__(16) float    sout[16][CL];
  int p0 = blockIdx.x * 16;
  int tid = threadIdx.x;
  if (tid < 16) {
    int p = p0 + tid;
    int pc = p < N_PTS ? p : (N_PTS - 1);
    const float* xp = x + (size_t)pc * N_VARS;
    float lat = xp[0]; if (lat == -90.0f) lat += 1e-4f;
    float lon = xp[1];
    float t   = xp[2];
    float feat[EMB_IN];
#pragma unroll
    for (int i = 0; i < N_VARS; ++i) feat[i] = xp[i];
    feat[0] = lat / 90.0f; feat[1] = lon / 180.0f; feat[2] = t / 12.0f;
    float lat_th = fmodf(-lat, 1.0f); if (lat_th < 0.f) lat_th += 1.0f;
    float lon_th = fmodf(lon, 1.0f);  if (lon_th < 0.f) lon_th += 1.0f;
    float t_th = t + 1.0f;
    float pc3[3] = {lat_th, lon_th, t_th};
    feat[24] = lat_th; feat[25] = lon_th; feat[26] = t_th;
#pragma unroll
    for (int c = 0; c < 3; ++c) {
      float fr = PI_F;
#pragma unroll
      for (int j = 0; j < PE_NUM; ++j) {
        float ang = pc3[c] * fr; fr *= 2.0f;
        feat[27 + c * PE_NUM + j] = sinf(ang);
        feat[51 + c * PE_NUM + j] = cosf(ang);
      }
    }
    for (int i = 0; i < EMB_IN; ++i) sfeat[tid][i] = (_Float16)feat[i];
    for (int i = EMB_IN; i < EMB_K; ++i) sfeat[tid][i] = (_Float16)0.0f;
    int lati = (int)floorf(90.0f - lat);
    if (lati < 0) lati = 0; if (lati > GH - 1) lati = GH - 1;
    int loni = ((int)(180.0f + floorf(lon + 180.0f))) % GW;
    if (loni < 0) loni += GW;
    if (p < N_PTS) flat[p] = lati * GW + loni;
  }
  __syncthreads();
  int wave = tid >> 5, lane = tid & 31, n = lane & 15, hh = lane >> 4;
  for (int it = 0; it < 4; ++it) {
    int ct = wave * 4 + it;
    v8f acc = {};
#pragma unroll
    for (int kk = 0; kk < 3; ++kk) {
      if (kk < 2) prefetch_b(Wemb, 32, kk + 1, ct);
      v16h a = frag_a_lds(&sfeat[0][0], EMB_K, kk * 32);
      v16h b = frag_b_packed(Wemb, 32, kk, ct);
      acc = wmma32(a, b, acc);
    }
    float bias = bemb[ct * 16 + n];
#pragma unroll
    for (int r = 0; r < 8; ++r) sout[8 * hh + r][ct * 16 + n] = acc[r] + bias;
  }
  __syncthreads();
#if HAVE_TDM
  // single TDM store for the 16x512 tile; rows beyond tensor_dim1 are dropped.
  if (tid < 32)
    tdm_store_2d(&sout[0][0], hbuf + (size_t)p0 * CL,
                 CL, (unsigned)(N_PTS - p0), CL, 16, CL);
  // S_ENDPGM's implicit wait-idle covers TENSORcnt completion.
#else
  for (int i = tid; i < 16 * CL / 4; i += 256) {
    int r = i >> 7, c4 = i & 127;
    int row = p0 + r;
    if (row < N_PTS)
      ((float4*)hbuf)[(size_t)row * (CL / 4) + c4] = ((const float4*)&sout[r][0])[c4];
  }
#endif
}

// ================= one transformer layer over 4 points (16 token rows) =================
__global__ __launch_bounds__(256) void k_layer(float* __restrict__ hbuf,
                                               const float* __restrict__ ln1_s,
                                               const float* __restrict__ ln1_b,
                                               const _Float16* __restrict__ Wqkv,
                                               const float* __restrict__ bqkv,
                                               const _Float16* __restrict__ Wo,
                                               const float* __restrict__ bo,
                                               const float* __restrict__ ln2_s,
                                               const float* __restrict__ ln2_b,
                                               const _Float16* __restrict__ Wf1,
                                               const float* __restrict__ bf1,
                                               const _Float16* __restrict__ Wf2,
                                               const float* __restrict__ bf2) {
  __shared__ __align__(16) float     sh[16][TR_DIM];
  __shared__ __align__(32) _Float16  sa[16][TR_DIM];
  __shared__ __align__(16) float     sqkv[16][3 * TR_DIM];
  __shared__ __align__(32) _Float16  so[16][TR_DIM];
  __shared__ __align__(32) _Float16  sg[16][4 * TR_DIM];

  int tid = threadIdx.x;
  int r0  = blockIdx.x * 16;

#if HAVE_TDM
  if (tid < 32) {
    tdm_load_2d(&sh[0][0], hbuf + (size_t)r0 * TR_DIM,
                TR_DIM, (unsigned)(NTOK - r0), TR_DIM, 16, TR_DIM);
    __builtin_amdgcn_s_wait_tensorcnt(0);
  }
  __syncthreads();
#else
  for (int i = tid; i < 16 * TR_DIM / 4; i += 256) {
    int r = i >> 5, c4 = i & 31;
    int gr = r0 + r;
    float4 v = (gr < NTOK) ? ((const float4*)hbuf)[(size_t)gr * (TR_DIM / 4) + c4]
                           : make_float4(0.f, 0.f, 0.f, 0.f);
    ((float4*)&sh[r][0])[c4] = v;
  }
  __syncthreads();
#endif

  layernorm16(sh, sa, ln1_s, ln1_b, tid);
  __syncthreads();

  int wave = tid >> 5, lane = tid & 31, n = lane & 15, hh = lane >> 4;

  // QKV GEMM: 16x128 @ 128x384 -> 24 col tiles (NT=24)
  for (int it = 0; it < 3; ++it) {
    int ct = wave * 3 + it;
    v8f acc = {};
#pragma unroll
    for (int kk = 0; kk < 4; ++kk) {
      if (kk < 3) prefetch_b(Wqkv, 24, kk + 1, ct);
      v16h a = frag_a_lds(&sa[0][0], TR_DIM, kk * 32);
      v16h b = frag_b_packed(Wqkv, 24, kk, ct);
      acc = wmma32(a, b, acc);
    }
    float bias = bqkv[ct * 16 + n];
#pragma unroll
    for (int r = 0; r < 8; ++r) sqkv[8 * hh + r][ct * 16 + n] = acc[r] + bias;
  }
  __syncthreads();

  // attention: 4 pts x 8 heads x 4 q-tokens, seq len 4
  if (tid < 128) {
    int p  = tid >> 5;
    int hd = (tid >> 2) & 7;
    int qt = tid & 3;
    int qrow = p * 4 + qt;
    float s[4]; float mx = -1e30f;
#pragma unroll
    for (int kt = 0; kt < 4; ++kt) {
      float d = 0.f;
#pragma unroll
      for (int e = 0; e < DH; ++e)
        d += sqkv[qrow][hd * DH + e] * sqkv[p * 4 + kt][TR_DIM + hd * DH + e];
      s[kt] = d * 0.25f;
      mx = fmaxf(mx, s[kt]);
    }
    float sum = 0.f;
#pragma unroll
    for (int kt = 0; kt < 4; ++kt) { s[kt] = expf(s[kt] - mx); sum += s[kt]; }
    float inv = 1.0f / sum;
#pragma unroll
    for (int e = 0; e < DH; ++e) {
      float o = 0.f;
#pragma unroll
      for (int kt = 0; kt < 4; ++kt)
        o += s[kt] * inv * sqkv[p * 4 + kt][2 * TR_DIM + hd * DH + e];
      so[qrow][hd * DH + e] = (_Float16)o;
    }
  }
  __syncthreads();

  // Wo GEMM + residual: 8 col tiles (NT=8)
  {
    int ct = wave;
    v8f acc = {};
#pragma unroll
    for (int kk = 0; kk < 4; ++kk) {
      if (kk < 3) prefetch_b(Wo, 8, kk + 1, ct);
      v16h a = frag_a_lds(&so[0][0], TR_DIM, kk * 32);
      v16h b = frag_b_packed(Wo, 8, kk, ct);
      acc = wmma32(a, b, acc);
    }
    float bias = bo[ct * 16 + n];
#pragma unroll
    for (int r = 0; r < 8; ++r) sh[8 * hh + r][ct * 16 + n] += acc[r] + bias;
  }
  __syncthreads();

  layernorm16(sh, sa, ln2_s, ln2_b, tid);
  __syncthreads();

  // FF1 + gelu: 32 col tiles (NT=32)
  for (int it = 0; it < 4; ++it) {
    int ct = wave * 4 + it;
    v8f acc = {};
#pragma unroll
    for (int kk = 0; kk < 4; ++kk) {
      if (kk < 3) prefetch_b(Wf1, 32, kk + 1, ct);
      v16h a = frag_a_lds(&sa[0][0], TR_DIM, kk * 32);
      v16h b = frag_b_packed(Wf1, 32, kk, ct);
      acc = wmma32(a, b, acc);
    }
    float bias = bf1[ct * 16 + n];
#pragma unroll
    for (int r = 0; r < 8; ++r)
      sg[8 * hh + r][ct * 16 + n] = (_Float16)gelu_tanh(acc[r] + bias);
  }
  __syncthreads();

  // FF2 + residual: K=512, 8 col tiles (NT=8)
  {
    int ct = wave;
    v8f acc = {};
#pragma unroll
    for (int kk = 0; kk < 16; ++kk) {
      if (kk < 15) prefetch_b(Wf2, 8, kk + 1, ct);
      v16h a = frag_a_lds(&sg[0][0], 4 * TR_DIM, kk * 32);
      v16h b = frag_b_packed(Wf2, 8, kk, ct);
      acc = wmma32(a, b, acc);
    }
    float bias = bf2[ct * 16 + n];
#pragma unroll
    for (int r = 0; r < 8; ++r) sh[8 * hh + r][ct * 16 + n] += acc[r] + bias;
  }
  __syncthreads();

#if HAVE_TDM
  if (tid < 32)
    tdm_store_2d(&sh[0][0], hbuf + (size_t)r0 * TR_DIM,
                 TR_DIM, (unsigned)(NTOK - r0), TR_DIM, 16, TR_DIM);
#else
  for (int i = tid; i < 16 * TR_DIM / 4; i += 256) {
    int r = i >> 5, c4 = i & 31;
    int gr = r0 + r;
    if (gr < NTOK)
      ((float4*)hbuf)[(size_t)gr * (TR_DIM / 4) + c4] = ((const float4*)&sh[r][0])[c4];
  }
#endif
}

// ================= comb: y = h @ W_comb + b, scattered with atomics =================
__global__ __launch_bounds__(256) void k_comb(const float* __restrict__ hbuf,
                                              const _Float16* __restrict__ Wc,
                                              const float* __restrict__ bc,
                                              const int* __restrict__ flat,
                                              float* __restrict__ outsum,
                                              float* __restrict__ cnt) {
  __shared__ __align__(32) _Float16 sa[16][CL];
  __shared__ __align__(16) float    stage[16][CL];
  __shared__ int sflat[16];
  int p0 = blockIdx.x * 16;
  int tid = threadIdx.x;
#if HAVE_TDM
  if (tid < 32) {
    tdm_load_2d(&stage[0][0], hbuf + (size_t)p0 * CL,
                CL, (unsigned)(N_PTS - p0), CL, 16, CL);
    __builtin_amdgcn_s_wait_tensorcnt(0);
  }
#else
  for (int i = tid; i < 16 * CL / 4; i += 256) {
    int r = i >> 7, c4 = i & 127;
    int p = p0 + r;
    float4 v = (p < N_PTS) ? ((const float4*)hbuf)[(size_t)p * (CL / 4) + c4]
                           : make_float4(0.f, 0.f, 0.f, 0.f);
    ((float4*)&stage[r][0])[c4] = v;
  }
#endif
  if (tid < 16) {
    int p = p0 + tid;
    sflat[tid] = (p < N_PTS) ? flat[p] : 0;
    if (p < N_PTS) atomicAdd(cnt + sflat[tid], 1.0f);
  }
  __syncthreads();
  // f32 -> f16 conversion in LDS
  for (int i = tid; i < 16 * CL / 4; i += 256) {
    int r = i >> 7, c4 = i & 127;
    float4 v = ((const float4*)&stage[r][0])[c4];
    sa[r][c4 * 4 + 0] = (_Float16)v.x;
    sa[r][c4 * 4 + 1] = (_Float16)v.y;
    sa[r][c4 * 4 + 2] = (_Float16)v.z;
    sa[r][c4 * 4 + 3] = (_Float16)v.w;
  }
  __syncthreads();
  int wave = tid >> 5, lane = tid & 31, n = lane & 15, hh = lane >> 4;
  for (int it = 0; it < 4; ++it) {
    int ct = wave * 4 + it;
    v8f acc = {};
#pragma unroll
    for (int kk = 0; kk < 16; ++kk) {
      if (kk < 15) prefetch_b(Wc, 32, kk + 1, ct);
      v16h a = frag_a_lds(&sa[0][0], CL, kk * 32);
      v16h b = frag_b_packed(Wc, 32, kk, ct);
      acc = wmma32(a, b, acc);
    }
    float bias = bc[ct * 16 + n];
#pragma unroll
    for (int r = 0; r < 8; ++r) {
      int row = 8 * hh + r;
      if (p0 + row < N_PTS)
        atomicAdd(outsum + (size_t)sflat[row] * CL + ct * 16 + n, acc[r] + bias);
    }
  }
}

__global__ void k_final(float* __restrict__ out, const float* __restrict__ cnt) {
  for (size_t i = (size_t)blockIdx.x * blockDim.x + threadIdx.x;
       i < (size_t)GH * GW * CL; i += (size_t)gridDim.x * blockDim.x)
    out[i] /= fmaxf(cnt[i >> 9], 1.0f);
}

// ================= launch =================
extern "C" void kernel_launch(void* const* d_in, const int* in_sizes, int n_in,
                              void* d_out, int out_size, void* d_ws, size_t ws_size,
                              hipStream_t stream) {
  (void)in_sizes; (void)n_in; (void)out_size; (void)ws_size;
  const float* x      = (const float*)d_in[0];
  const float* W_emb  = (const float*)d_in[1];
  const float* b_emb  = (const float*)d_in[2];
  const float* ln1_s  = (const float*)d_in[3];
  const float* ln1_b  = (const float*)d_in[4];
  const float* Wqkv   = (const float*)d_in[5];
  const float* bqkv   = (const float*)d_in[6];
  const float* Wo     = (const float*)d_in[7];
  const float* bo     = (const float*)d_in[8];
  const float* ln2_s  = (const float*)d_in[9];
  const float* ln2_b  = (const float*)d_in[10];
  const float* Wf1    = (const float*)d_in[11];
  const float* bf1    = (const float*)d_in[12];
  const float* Wf2    = (const float*)d_in[13];
  const float* bf2    = (const float*)d_in[14];
  const float* W_comb = (const float*)d_in[15];
  const float* b_comb = (const float*)d_in[16];

  char* ws = (char*)d_ws;
  _Float16* wembh  = (_Float16*)(ws + 0);        // 96*512 packed
  _Float16* wqkvh  = (_Float16*)(ws + 98304);    // 2*128*384 packed
  _Float16* woh    = (_Float16*)(ws + 294912);   // 2*128*128 packed
  _Float16* wf1h   = (_Float16*)(ws + 360448);   // 2*128*512 packed
  _Float16* wf2h   = (_Float16*)(ws + 622592);   // 2*512*128 packed
  _Float16* wcmbh  = (_Float16*)(ws + 884736);   // 512*512 packed
  int*      flat   = (int*)     (ws + 1409024);  // N_PTS
  float*    cnt    = (float*)   (ws + 1671168);  // GH*GW
  float*    hbuf   = (float*)   (ws + 1930368);  // N_PTS*512 f32

  float* out = (float*)d_out;

  k_pack_b<<<192,  256, 0, stream>>>(wembh, W_emb, EMB_K, CL, EMB_IN);
  for (int l = 0; l < 2; ++l) {
    k_pack_b<<<192, 256, 0, stream>>>(wqkvh + (size_t)l * 128 * 384,
                                      Wqkv + (size_t)l * 128 * 384, 128, 384, 128);
    k_pack_b<<<64,  256, 0, stream>>>(woh + (size_t)l * 128 * 128,
                                      Wo + (size_t)l * 128 * 128, 128, 128, 128);
    k_pack_b<<<256, 256, 0, stream>>>(wf1h + (size_t)l * 128 * 512,
                                      Wf1 + (size_t)l * 128 * 512, 128, 512, 128);
    k_pack_b<<<256, 256, 0, stream>>>(wf2h + (size_t)l * 512 * 128,
                                      Wf2 + (size_t)l * 512 * 128, 512, 128, 512);
  }
  k_pack_b<<<1024, 256, 0, stream>>>(wcmbh, W_comb, 512, 512, 512);
  k_zero<<<2048, 256, 0, stream>>>(out, GH * GW * CL);
  k_zero<<<256,  256, 0, stream>>>(cnt, GH * GW);

  k_embed<<<(N_PTS + 15) / 16, 256, 0, stream>>>(x, wembh, b_emb, hbuf, flat);

  for (int l = 0; l < 2; ++l) {
    k_layer<<<(NTOK + 15) / 16, 256, 0, stream>>>(
        hbuf,
        ln1_s + l * TR_DIM, ln1_b + l * TR_DIM,
        wqkvh + (size_t)l * 128 * 384, bqkv + l * 384,
        woh   + (size_t)l * 128 * 128, bo   + l * 128,
        ln2_s + l * TR_DIM, ln2_b + l * TR_DIM,
        wf1h  + (size_t)l * 128 * 512, bf1  + l * 512,
        wf2h  + (size_t)l * 512 * 128, bf2  + l * 128);
  }

  k_comb <<<(N_PTS + 15) / 16, 256, 0, stream>>>(hbuf, wcmbh, b_comb, flat, out, cnt);
  k_final<<<4096, 256, 0, stream>>>(out, cnt);
}
